// SwitchMoE_80083960201401
// MI455X (gfx1250) — matrix operand order repaired and checked by reading, not compiled, not run on hardware
//
#include <hip/hip_runtime.h>
#include <hip/hip_bf16.h>
#include <cmath>

typedef __attribute__((ext_vector_type(16))) __bf16 v16bf;
typedef __attribute__((ext_vector_type(8)))  float  v8f;

struct __align__(8) bf16x4 { __bf16 v[4]; };

#define T_TOK   2048
#define C_IN    1024
#define F_HID   4096
#define NEXP    8
#define GDIM    16

#define APAD 72    // A row stride (bf16): 144B, 8B aligned, conflict-free across 16 rows
#define BPAD 132   // B row stride (bf16): 264B, 8B aligned, 66-dword stride conflict-free

// ---------------- workspace layout (bytes) ----------------
#define WS_COUNTS  0
#define WS_EIDX    1024
#define WS_SCORE   9216
#define WS_BUCKET  17408
#define WS_H       131072   // bf16[2048][4096] = 16 MB

// ================= gate: reduce->cosine->softmax top-1 =================
__global__ void moe_gate_kernel(const float* __restrict__ x,
                                const float* __restrict__ wg_red,
                                const float* __restrict__ wg,
                                int* __restrict__ eidx,
                                float* __restrict__ score) {
    const int wave = threadIdx.x >> 5;
    const int lane = threadIdx.x & 31;
    const int t = blockIdx.x * 8 + wave;
    if (t >= T_TOK) return;
    const float* xt = x + (size_t)t * C_IN;

    float acc[GDIM];
#pragma unroll
    for (int g = 0; g < GDIM; ++g) acc[g] = 0.f;
    for (int i = lane; i < C_IN; i += 32) {
        const float xv = xt[i];
#pragma unroll
        for (int g = 0; g < GDIM; ++g) acc[g] += xv * wg_red[g * C_IN + i];
    }
#pragma unroll
    for (int g = 0; g < GDIM; ++g) {
#pragma unroll
        for (int off = 16; off > 0; off >>= 1)
            acc[g] += __shfl_xor(acc[g], off, 32);
    }
    float logit = -1e30f;
    if (lane < NEXP) {
        float n2 = 0.f, dot = 0.f;
#pragma unroll
        for (int g = 0; g < GDIM; ++g) {
            const float w = wg[lane * GDIM + g];
            n2  += w * w;
            dot += acc[g] * w;
        }
        logit = dot / fmaxf(sqrtf(n2), 1e-4f);
    }
    float l[NEXP];
#pragma unroll
    for (int e = 0; e < NEXP; ++e) l[e] = __shfl(logit, e, 32);
    if (lane == 0) {
        float mx = l[0]; int am = 0;
        for (int e = 1; e < NEXP; ++e) if (l[e] > mx) { mx = l[e]; am = e; }
        float s = 0.f;
        for (int e = 0; e < NEXP; ++e) s += expf(l[e] - mx);
        eidx[t]  = am;
        score[t] = 1.0f / s;
    }
}

// ================= bucket: compact token ids per expert =================
__global__ void moe_bucket_kernel(const int* __restrict__ eidx,
                                  int* __restrict__ counts,
                                  int* __restrict__ bucket) {
    const int t = blockIdx.x * blockDim.x + threadIdx.x;
    if (t >= T_TOK) return;
    const int e = eidx[t];
    const int pos = atomicAdd(&counts[e], 1);
    bucket[e * T_TOK + pos] = t;
}

// ---------------- fragment helpers ----------------
__device__ __forceinline__ v16bf load_a_frag(const __bf16* sA, int row, int lh, int koff) {
    v16bf a;
#pragma unroll
    for (int j = 0; j < 16; ++j) {
        const int k = ((j < 8) ? (j + 8 * lh) : (j + 8 + 8 * lh)) + koff;
        a[j] = sA[row * APAD + k];
    }
    return a;
}
__device__ __forceinline__ v16bf load_b_frag(const __bf16* sB, int lane, int nw, int koff) {
    v16bf b;
#pragma unroll
    for (int j = 0; j < 16; ++j) {
        const int k = ((j < 8) ? (j + 8 * (lane >> 4)) : (j + 8 + 8 * (lane >> 4))) + koff;
        b[j] = sB[k * BPAD + nw + (lane & 15)];
    }
    return b;
}
__device__ __forceinline__ bf16x4 pack4(const float4 f) {
    bf16x4 p;
    p.v[0] = (__bf16)f.x; p.v[1] = (__bf16)f.y;
    p.v[2] = (__bf16)f.z; p.v[3] = (__bf16)f.w;
    return p;
}

// ================= up-proj: H = gelu(x @ W1[e] + b1[e]) =================
// grid: (F_HID/128, T_TOK/32, NEXP), block 256; double-buffered LDS pipeline
__global__ void moe_up_kernel(const float* __restrict__ x,
                              const float* __restrict__ w1,
                              const float* __restrict__ b1,
                              const int* __restrict__ counts,
                              const int* __restrict__ bucket,
                              __bf16* __restrict__ H) {
    __shared__ __bf16 sA[2][32 * APAD];
    __shared__ __bf16 sB[2][64 * BPAD];
    __shared__ int s_tok[32];

    const int e  = blockIdx.z;
    const int m0 = blockIdx.y * 32;
    const int n0 = blockIdx.x * 128;
    const int count = counts[e];
    if (m0 >= count) return;
    const int rows = min(32, count - m0);

    const int tid  = threadIdx.x;
    const int wave = tid >> 5, lane = tid & 31;
    const int lh = lane >> 4, lm = lane & 15;
    const int nw = wave * 16;

    if (tid < 32) s_tok[tid] = (tid < rows) ? bucket[e * T_TOK + m0 + tid] : 0;
    __syncthreads();

    // A staging: rows ra, ra+16 (token-clamped -> unconditional loads); quad kqa
    const int ra  = tid >> 4;            // 0..15
    const int kqa = (tid & 15) * 4;      // 0..60
    const float* ap0 = x + (size_t)s_tok[ra] * C_IN + kqa;
    const float* ap1 = x + (size_t)s_tok[ra + 16] * C_IN + kqa;

    // B staging: 8 rows k = kb+8i, quad nq
    const int kb = tid >> 5;             // 0..7
    const int nq = (tid & 31) * 4;       // 0..124
    const float* w1e = w1 + (size_t)e * C_IN * F_HID + n0 + nq;

    float4 fa0, fa1, fb[8];
    auto load_tile = [&](int k0) {
        fa0 = *(const float4*)(ap0 + k0);
        fa1 = *(const float4*)(ap1 + k0);
#pragma unroll
        for (int i = 0; i < 8; ++i)
            fb[i] = *(const float4*)(w1e + (size_t)(k0 + kb + 8 * i) * F_HID);
    };
    auto store_tile = [&](int buf) {
        *(bf16x4*)&sA[buf][ra * APAD + kqa]        = pack4(fa0);
        *(bf16x4*)&sA[buf][(ra + 16) * APAD + kqa] = pack4(fa1);
#pragma unroll
        for (int i = 0; i < 8; ++i)
            *(bf16x4*)&sB[buf][(kb + 8 * i) * BPAD + nq] = pack4(fb[i]);
    };

    v8f c0 = {}, c1 = {};
    constexpr int NT = C_IN / 64;

    load_tile(0);
    store_tile(0);
    __syncthreads();

#pragma unroll 2
    for (int i = 0; i < NT; ++i) {
        const int cur = i & 1;
        const bool more = (i + 1) < NT;
        if (more) load_tile((i + 1) * 64);     // global loads overlap compute below
#pragma unroll
        for (int koff = 0; koff < 64; koff += 32) {
            const v16bf bfr = load_b_frag(sB[cur], lane, nw, koff);
            const v16bf a0  = load_a_frag(sA[cur], lm,      lh, koff);
            const v16bf a1  = load_a_frag(sA[cur], lm + 16, lh, koff);
            c0 = __builtin_amdgcn_wmma_f32_16x16x32_bf16(false, a0, false, bfr, (short)0, c0, false, false);
            c1 = __builtin_amdgcn_wmma_f32_16x16x32_bf16(false, a1, false, bfr, (short)0, c1, false, false);
        }
        if (more) store_tile(cur ^ 1);
        __syncthreads();
    }

    const int gn = n0 + nw + lm;
    const float bias = b1[e * F_HID + gn];
#pragma unroll
    for (int sm = 0; sm < 2; ++sm) {
        const v8f& c = sm ? c1 : c0;
#pragma unroll
        for (int v = 0; v < 8; ++v) {
            const int m = sm * 16 + v + 8 * lh;
            if (m < rows) {
                float h = c[v] + bias;
                h = 0.5f * h * (1.0f + erff(h * 0.70710678118654752f));
                H[(size_t)s_tok[m] * F_HID + gn] = (__bf16)h;
            }
        }
    }
}

// ================= down-proj: out = (H @ W2[e] + b2[e]) * score =================
// grid: (C_IN/128, T_TOK/32, NEXP), block 256; double-buffered LDS pipeline
__global__ void moe_down_kernel(const __bf16* __restrict__ H,
                                const float* __restrict__ w2,
                                const float* __restrict__ b2,
                                const int* __restrict__ counts,
                                const int* __restrict__ bucket,
                                const float* __restrict__ score,
                                float* __restrict__ out) {
    __shared__ __bf16 sA[2][32 * APAD];
    __shared__ __bf16 sB[2][64 * BPAD];
    __shared__ int s_tok[32];

    const int e  = blockIdx.z;
    const int m0 = blockIdx.y * 32;
    const int n0 = blockIdx.x * 128;
    const int count = counts[e];
    if (m0 >= count) return;
    const int rows = min(32, count - m0);

    const int tid  = threadIdx.x;
    const int wave = tid >> 5, lane = tid & 31;
    const int lh = lane >> 4, lm = lane & 15;
    const int nw = wave * 16;

    if (tid < 32) s_tok[tid] = (tid < rows) ? bucket[e * T_TOK + m0 + tid] : 0;
    __syncthreads();

    // A staging: bf16 source; thread copies 8 bf16 (uint4), token-clamped
    const int ra = tid >> 3;             // 0..31
    const int ka = (tid & 7) * 8;        // 0..56
    const __bf16* hp = H + (size_t)s_tok[ra] * F_HID + ka;

    const int kb = tid >> 5;             // 0..7
    const int nq = (tid & 31) * 4;       // 0..124
    const float* w2e = w2 + (size_t)e * F_HID * C_IN + n0 + nq;

    uint4 ua;
    float4 fb[8];
    auto load_tile = [&](int k0) {
        ua = *(const uint4*)(hp + k0);
#pragma unroll
        for (int i = 0; i < 8; ++i)
            fb[i] = *(const float4*)(w2e + (size_t)(k0 + kb + 8 * i) * C_IN);
    };
    auto store_tile = [&](int buf) {
        *(uint4*)&sA[buf][ra * APAD + ka] = ua;
#pragma unroll
        for (int i = 0; i < 8; ++i)
            *(bf16x4*)&sB[buf][(kb + 8 * i) * BPAD + nq] = pack4(fb[i]);
    };

    v8f c0 = {}, c1 = {};
    constexpr int NT = F_HID / 64;

    load_tile(0);
    store_tile(0);
    __syncthreads();

#pragma unroll 2
    for (int i = 0; i < NT; ++i) {
        const int cur = i & 1;
        const bool more = (i + 1) < NT;
        if (more) load_tile((i + 1) * 64);
#pragma unroll
        for (int koff = 0; koff < 64; koff += 32) {
            const v16bf bfr = load_b_frag(sB[cur], lane, nw, koff);
            const v16bf a0  = load_a_frag(sA[cur], lm,      lh, koff);
            const v16bf a1  = load_a_frag(sA[cur], lm + 16, lh, koff);
            c0 = __builtin_amdgcn_wmma_f32_16x16x32_bf16(false, a0, false, bfr, (short)0, c0, false, false);
            c1 = __builtin_amdgcn_wmma_f32_16x16x32_bf16(false, a1, false, bfr, (short)0, c1, false, false);
        }
        if (more) store_tile(cur ^ 1);
        __syncthreads();
    }

    const int gn = n0 + nw + lm;
    const float bias = b2[e * C_IN + gn];
#pragma unroll
    for (int sm = 0; sm < 2; ++sm) {
        const v8f& c = sm ? c1 : c0;
#pragma unroll
        for (int v = 0; v < 8; ++v) {
            const int m = sm * 16 + v + 8 * lh;
            if (m < rows) {
                const int t = s_tok[m];
                out[(size_t)t * C_IN + gn] = (c[v] + bias) * score[t];
            }
        }
    }
}

extern "C" void kernel_launch(void* const* d_in, const int* in_sizes, int n_in,
                              void* d_out, int out_size, void* d_ws, size_t ws_size,
                              hipStream_t stream) {
    const float* x      = (const float*)d_in[0];
    const float* wg_red = (const float*)d_in[1];
    const float* wg     = (const float*)d_in[2];
    const float* w1     = (const float*)d_in[3];
    const float* b1     = (const float*)d_in[4];
    const float* w2     = (const float*)d_in[5];
    const float* b2     = (const float*)d_in[6];
    float* out = (float*)d_out;

    char* ws = (char*)d_ws;
    int*    counts = (int*)(ws + WS_COUNTS);
    int*    eidx   = (int*)(ws + WS_EIDX);
    float*  score  = (float*)(ws + WS_SCORE);
    int*    bucket = (int*)(ws + WS_BUCKET);
    __bf16* Hbuf   = (__bf16*)(ws + WS_H);

    hipMemsetAsync(counts, 0, NEXP * sizeof(int), stream);

    moe_gate_kernel<<<T_TOK / 8, 256, 0, stream>>>(x, wg_red, wg, eidx, score);
    moe_bucket_kernel<<<T_TOK / 256, 256, 0, stream>>>(eidx, counts, bucket);

    dim3 gUp(F_HID / 128, T_TOK / 32, NEXP);
    moe_up_kernel<<<gUp, 256, 0, stream>>>(x, w1, b1, counts, bucket, Hbuf);

    dim3 gDn(C_IN / 128, T_TOK / 32, NEXP);
    moe_down_kernel<<<gDn, 256, 0, stream>>>(Hbuf, w2, b2, counts, bucket, score, out);
}
